// SwinTransformerBlock3D_52252572123370
// MI455X (gfx1250) — compile-verified, matrix operand order checked by reading, not run on hardware
//
#include <hip/hip_runtime.h>
#include <hip/hip_bf16.h>
#include <math.h>

// ---------------------------------------------------------------------------
// Types for CDNA5 WMMA
// ---------------------------------------------------------------------------
typedef __attribute__((ext_vector_type(16))) __bf16 v16bf;
typedef __attribute__((ext_vector_type(8)))  float  v8f;

union FragU {
    v16bf v;
    struct { uint4 u0; uint4 u1; } q;
};

// CDNA5 16-bit A/B fragment loader (two contiguous 16-byte chunks per lane):
//   lanes 0-15 : row = row0+lane,    K chunks {0..7} and {16..23}
//   lanes 16-31: row = row0+lane-16, K chunks {8..15} and {24..31}
__device__ __forceinline__ v16bf load_frag(const __bf16* base, int ld, int row0, int col0) {
    int lane = threadIdx.x & 31;
    const __bf16* p = base + (size_t)(row0 + (lane & 15)) * (size_t)ld
                           + (size_t)(col0 + ((lane >> 4) << 3));
    FragU f;
    f.q.u0 = *reinterpret_cast<const uint4*>(p);
    f.q.u1 = *reinterpret_cast<const uint4*>(p + 16);
    return f.v;
}

__device__ __forceinline__ v8f wmma_bf16(v16bf a, v16bf b, v8f c) {
    return __builtin_amdgcn_wmma_f32_16x16x32_bf16(false, a, false, b, (short)0, c, false, false);
}

// ---------------------------------------------------------------------------
// Async global -> LDS staging (CDNA5 ASYNCcnt path) with safe fallback
// ---------------------------------------------------------------------------
#if defined(__gfx1250__) && __has_builtin(__builtin_amdgcn_global_load_async_to_lds_b128)
#define USE_ASYNC_LDS 1
#else
#define USE_ASYNC_LDS 0
#endif

#if USE_ASYNC_LDS
typedef int vint4 __attribute__((vector_size(16)));
typedef __attribute__((address_space(1))) vint4* g_v4i_ptr;
typedef __attribute__((address_space(3))) vint4* l_v4i_ptr;
__device__ __forceinline__ void cp16(const __bf16* g, __bf16* l) {
    __builtin_amdgcn_global_load_async_to_lds_b128((g_v4i_ptr)(g), (l_v4i_ptr)(l), 0, 0);
}
__device__ __forceinline__ void wait_async_le4() { asm volatile("s_wait_asynccnt 0x4" ::: "memory"); }
__device__ __forceinline__ void wait_async_0()  { asm volatile("s_wait_asynccnt 0x0" ::: "memory"); }
#else
__device__ __forceinline__ void cp16(const __bf16* g, __bf16* l) {
    *reinterpret_cast<uint4*>(l) = *reinterpret_cast<const uint4*>(g);
}
__device__ __forceinline__ void wait_async_le4() {}
__device__ __forceinline__ void wait_async_0()  {}
#endif

// ---------------------------------------------------------------------------
// Problem constants
// ---------------------------------------------------------------------------
#define BATCH   4
#define DD      8
#define HH      56
#define WW2     56
#define C       384
#define CH      1536
#define HEADS   12
#define HD      32
#define NTOK    98
#define NPAD    112
#define NWIN    1024
#define MROWS   (NWIN * NPAD)            // 114688
#define TOKENS  (BATCH * DD * HH * WW2)  // 100352
#define QKVC    (3 * C)                  // 1152
#define SCALE_Q 0.17677669529663689f     // 32^-0.5

// GEMM tiling
#define BM   128
#define BN   128
#define BK   32
#define LDP  40   // padded LDS row stride (bf16): 20*r mod 64 -> 16 distinct banks

// ---------------------------------------------------------------------------
// Shared GEMM core: block tile 128x128, 8 waves (4M x 2N), wave tile 32x64,
// double-buffered LDS staging. A:(M,K) row-major, W:(N,K) row-major.
// ---------------------------------------------------------------------------
__device__ __forceinline__ void gemm_core(const __bf16* __restrict__ A,
                                          const __bf16* __restrict__ W,
                                          int K, int m0blk, int n0blk,
                                          v8f acc[2][4]) {
    __shared__ __bf16 sA[2][BM * LDP];
    __shared__ __bf16 sB[2][BN * LDP];

    int tid = threadIdx.x;
    int wave = tid >> 5;
    int wM = wave >> 1;     // 0..3
    int wN = wave & 1;      // 0..1

    auto stage = [&](int s, int k0) {
#pragma unroll
        for (int c = 0; c < 2; ++c) {
            int ch = tid + c * 256;           // 512 chunks of 8 bf16 per matrix
            int row = ch >> 2, q = (ch & 3) << 3;
            cp16(A + (size_t)(m0blk + row) * K + k0 + q, &sA[s][row * LDP + q]);
            cp16(W + (size_t)(n0blk + row) * K + k0 + q, &sB[s][row * LDP + q]);
        }
    };

    int steps = K / BK;
    stage(0, 0);
    for (int ks = 0; ks < steps; ++ks) {
        int s = ks & 1;
        if (ks + 1 < steps) { stage(s ^ 1, (ks + 1) * BK); wait_async_le4(); }
        else                { wait_async_0(); }
        __syncthreads();
        v16bf af0 = load_frag(sA[s], LDP, wM * 32, 0);
        v16bf af1 = load_frag(sA[s], LDP, wM * 32 + 16, 0);
#pragma unroll
        for (int t = 0; t < 4; ++t) {
            v16bf bf = load_frag(sB[s], LDP, wN * 64 + t * 16, 0);
            acc[0][t] = wmma_bf16(af0, bf, acc[0][t]);
            acc[1][t] = wmma_bf16(af1, bf, acc[1][t]);
        }
        __syncthreads();
    }
}

// ---------------------------------------------------------------------------
// Kernel: fp32 -> bf16 convert (grid-stride)
// ---------------------------------------------------------------------------
__global__ void k_convert_bf16(const float* __restrict__ in, __bf16* __restrict__ out, int n) {
    for (int i = blockIdx.x * blockDim.x + threadIdx.x; i < n; i += gridDim.x * blockDim.x)
        out[i] = (__bf16)in[i];
}

// ---------------------------------------------------------------------------
// Kernel: padded relative-position bias (12, 112, 112); cols>=98 -> -1e30
// ---------------------------------------------------------------------------
__global__ void k_bias_pad(const float* __restrict__ rpb, float* __restrict__ bias) {
    int idx = blockIdx.x * blockDim.x + threadIdx.x;
    if (idx >= HEADS * NPAD * NPAD) return;
    int h = idx / (NPAD * NPAD);
    int r = idx % (NPAD * NPAD);
    int i = r / NPAD, j = r % NPAD;
    float v;
    if (j >= NTOK)      v = -1e30f;
    else if (i >= NTOK) v = 0.0f;
    else {
        int di = i / 49, hi = (i / 7) % 7, wi = i % 7;
        int dj = j / 49, hj = (j / 7) % 7, wj = j % 7;
        int rel = (di - dj + 1) * 169 + (hi - hj + 6) * 13 + (wi - wj + 6);
        v = rpb[rel * HEADS + h];
    }
    bias[idx] = v;
}

// ---------------------------------------------------------------------------
// Kernel: LN1 + cyclic shift + window partition -> Xw bf16 (MROWS, C)
// ---------------------------------------------------------------------------
__global__ void k_ln1_partition(const float* __restrict__ x,
                                const float* __restrict__ g, const float* __restrict__ b,
                                __bf16* __restrict__ Xw) {
    int r = blockIdx.x * 8 + (threadIdx.x >> 5);
    int lane = threadIdx.x & 31;
    __bf16* outp = Xw + (size_t)r * C;
    int wb = r / NPAD, t = r % NPAD;
    if (t >= NTOK) {
#pragma unroll
        for (int i = 0; i < 12; ++i) outp[lane + 32 * i] = (__bf16)0.0f;
        return;
    }
    int ww = wb & 7, hw = (wb >> 3) & 7, dw = (wb >> 6) & 3, bt = wb >> 8;
    int td = t / 49, rem = t % 49, th = rem / 7, tw = rem % 7;
    int d = (dw * 2 + td + 1) & 7;
    int hh = hw * 7 + th + 3; if (hh >= HH) hh -= HH;
    int wpos = ww * 7 + tw + 3; if (wpos >= WW2) wpos -= WW2;
    const float* src = x + ((((size_t)bt * DD + d) * HH + hh) * WW2 + wpos) * C;

    float v[12], s = 0.f, s2 = 0.f;
#pragma unroll
    for (int i = 0; i < 12; ++i) { v[i] = src[lane + 32 * i]; s += v[i]; s2 += v[i] * v[i]; }
#pragma unroll
    for (int o = 16; o; o >>= 1) { s += __shfl_xor(s, o, 32); s2 += __shfl_xor(s2, o, 32); }
    float mu = s * (1.0f / C);
    float var = s2 * (1.0f / C) - mu * mu;
    float rs = rsqrtf(var + 1e-5f);
#pragma unroll
    for (int i = 0; i < 12; ++i) {
        int c = lane + 32 * i;
        outp[c] = (__bf16)((v[i] - mu) * rs * g[c] + b[c]);
    }
}

// ---------------------------------------------------------------------------
// Kernel: LN2 on residual output (d_out fp32) -> y bf16 (TOKENS, C)
// ---------------------------------------------------------------------------
__global__ void k_ln2(const float* __restrict__ xr,
                      const float* __restrict__ g, const float* __restrict__ b,
                      __bf16* __restrict__ y) {
    int r = blockIdx.x * 8 + (threadIdx.x >> 5);
    int lane = threadIdx.x & 31;
    const float* src = xr + (size_t)r * C;
    __bf16* outp = y + (size_t)r * C;
    float v[12], s = 0.f, s2 = 0.f;
#pragma unroll
    for (int i = 0; i < 12; ++i) { v[i] = src[lane + 32 * i]; s += v[i]; s2 += v[i] * v[i]; }
#pragma unroll
    for (int o = 16; o; o >>= 1) { s += __shfl_xor(s, o, 32); s2 += __shfl_xor(s2, o, 32); }
    float mu = s * (1.0f / C);
    float var = s2 * (1.0f / C) - mu * mu;
    float rs = rsqrtf(var + 1e-5f);
#pragma unroll
    for (int i = 0; i < 12; ++i) {
        int c = lane + 32 * i;
        outp[c] = (__bf16)((v[i] - mu) * rs * g[c] + b[c]);
    }
}

// ---------------------------------------------------------------------------
// GEMM kernels (grid = (M/128)*(N/128))
// ---------------------------------------------------------------------------
__global__ void k_gemm_qkv(const __bf16* __restrict__ A, const __bf16* __restrict__ W,
                           const float* __restrict__ bias, __bf16* __restrict__ out,
                           int M, int N, int K) {
    int nT = N >> 7;
    int m0 = (blockIdx.x / nT) * BM, n0 = (blockIdx.x % nT) * BN;
    v8f acc[2][4] = {};
    gemm_core(A, W, K, m0, n0, acc);
    int wave = threadIdx.x >> 5, lane = threadIdx.x & 31;
    int rowOff = (lane >> 4) << 3, colL = lane & 15;
#pragma unroll
    for (int i = 0; i < 2; ++i)
#pragma unroll
        for (int t = 0; t < 4; ++t)
#pragma unroll
            for (int j = 0; j < 8; ++j) {
                size_t r = m0 + (wave >> 1) * 32 + i * 16 + rowOff + j;
                int c = n0 + (wave & 1) * 64 + t * 16 + colL;
                out[r * (size_t)N + c] = (__bf16)(acc[i][t][j] + bias[c]);
            }
}

__global__ void k_gemm_proj(const __bf16* __restrict__ A, const __bf16* __restrict__ W,
                            const float* __restrict__ bias, const float* __restrict__ xres,
                            float* __restrict__ out, int M, int N, int K) {
    int nT = N >> 7;
    int m0 = (blockIdx.x / nT) * BM, n0 = (blockIdx.x % nT) * BN;
    v8f acc[2][4] = {};
    gemm_core(A, W, K, m0, n0, acc);
    int wave = threadIdx.x >> 5, lane = threadIdx.x & 31;
    int rowOff = (lane >> 4) << 3, colL = lane & 15;
#pragma unroll
    for (int i = 0; i < 2; ++i)
#pragma unroll
        for (int t = 0; t < 4; ++t)
#pragma unroll
            for (int j = 0; j < 8; ++j) {
                int r = m0 + (wave >> 1) * 32 + i * 16 + rowOff + j;
                int wb = r / NPAD, tk = r % NPAD;
                if (tk >= NTOK) continue;     // padded window rows
                int ww = wb & 7, hw = (wb >> 3) & 7, dw = (wb >> 6) & 3, bt = wb >> 8;
                int td = tk / 49, rem = tk % 49, th = rem / 7, tw = rem % 7;
                int d = (dw * 2 + td + 1) & 7;
                int hh = hw * 7 + th + 3; if (hh >= HH) hh -= HH;
                int wpos = ww * 7 + tw + 3; if (wpos >= WW2) wpos -= WW2;
                int c = n0 + (wave & 1) * 64 + t * 16 + colL;
                size_t idx = ((((size_t)bt * DD + d) * HH + hh) * WW2 + wpos) * C + c;
                out[idx] = xres[idx] + acc[i][t][j] + bias[c];
            }
}

__global__ void k_gemm_fc1(const __bf16* __restrict__ A, const __bf16* __restrict__ W,
                           const float* __restrict__ bias, __bf16* __restrict__ out,
                           int M, int N, int K) {
    int nT = N >> 7;
    int m0 = (blockIdx.x / nT) * BM, n0 = (blockIdx.x % nT) * BN;
    v8f acc[2][4] = {};
    gemm_core(A, W, K, m0, n0, acc);
    int wave = threadIdx.x >> 5, lane = threadIdx.x & 31;
    int rowOff = (lane >> 4) << 3, colL = lane & 15;
#pragma unroll
    for (int i = 0; i < 2; ++i)
#pragma unroll
        for (int t = 0; t < 4; ++t)
#pragma unroll
            for (int j = 0; j < 8; ++j) {
                size_t r = m0 + (wave >> 1) * 32 + i * 16 + rowOff + j;
                int c = n0 + (wave & 1) * 64 + t * 16 + colL;
                float v = acc[i][t][j] + bias[c];
                v = 0.5f * v * (1.0f + erff(v * 0.70710678118654752f));  // exact GELU
                out[r * (size_t)N + c] = (__bf16)v;
            }
}

__global__ void k_gemm_fc2(const __bf16* __restrict__ A, const __bf16* __restrict__ W,
                           const float* __restrict__ bias, float* __restrict__ out,
                           int M, int N, int K) {
    int nT = N >> 7;
    int m0 = (blockIdx.x / nT) * BM, n0 = (blockIdx.x % nT) * BN;
    v8f acc[2][4] = {};
    gemm_core(A, W, K, m0, n0, acc);
    int wave = threadIdx.x >> 5, lane = threadIdx.x & 31;
    int rowOff = (lane >> 4) << 3, colL = lane & 15;
#pragma unroll
    for (int i = 0; i < 2; ++i)
#pragma unroll
        for (int t = 0; t < 4; ++t)
#pragma unroll
            for (int j = 0; j < 8; ++j) {
                size_t r = m0 + (wave >> 1) * 32 + i * 16 + rowOff + j;
                int c = n0 + (wave & 1) * 64 + t * 16 + colL;
                out[r * (size_t)N + c] += acc[i][t][j] + bias[c];   // + x_after in d_out
            }
}

// ---------------------------------------------------------------------------
// Fused per-(window, head) attention.
//   LDS: Qs(112x32) Ks(112x32) Vt(32x128) S(112x112 f32) P(112x128) labels
// ---------------------------------------------------------------------------
__global__ void k_attention(const __bf16* __restrict__ QKV,
                            const float* __restrict__ biasPad,
                            __bf16* __restrict__ attnOut) {
    extern __shared__ __align__(16) char smem[];
    __bf16* Qs = (__bf16*)smem;                 // 112*32
    __bf16* Ks = Qs + NPAD * HD;                // 112*32
    __bf16* Vt = Ks + NPAD * HD;                // 32*128
    float*  S  = (float*)(Vt + HD * 128);       // 112*112
    __bf16* P  = (__bf16*)(S + NPAD * NPAD);    // 112*128
    int*    lab = (int*)(P + NPAD * 128);       // 112

    int wb = blockIdx.x / HEADS;
    int hh = blockIdx.x % HEADS;
    int tid = threadIdx.x;
    int wave = tid >> 5, lane = tid & 31;
    int ww = wb & 7, hw = (wb >> 3) & 7, dw = (wb >> 6) & 3;

    // stage Q and K (vectorized 16-byte chunks; 4 chunks per 32-wide head row)
    for (int idx = tid; idx < NPAD * 4; idx += 256) {
        int t = idx >> 2, q = (idx & 3) << 3;
        const __bf16* row = QKV + (size_t)(wb * NPAD + t) * QKVC;
        *reinterpret_cast<uint4*>(&Qs[t * HD + q]) =
            *reinterpret_cast<const uint4*>(&row[hh * HD + q]);
        *reinterpret_cast<uint4*>(&Ks[t * HD + q]) =
            *reinterpret_cast<const uint4*>(&row[C + hh * HD + q]);
    }
    // stage V transposed: Vt[c][t], K-dim padded 112 -> 128 with zeros
    for (int idx = tid; idx < HD * 128; idx += 256) {
        int c = idx >> 7, t = idx & 127;
        __bf16 v = (__bf16)0.0f;
        if (t < NPAD) v = QKV[(size_t)(wb * NPAD + t) * QKVC + 2 * C + hh * HD + c];
        Vt[idx] = v;
    }
    // shifted-window region labels
    if (tid < NPAD) {
        int t = tid;
        int td = t / 49, rem = t % 49, th = rem / 7, tw = rem % 7;
        int d = dw * 2 + td, h = hw * 7 + th, w = ww * 7 + tw;
        int dr = d < 6 ? 0 : (d < 7 ? 1 : 2);
        int hr = h < 49 ? 0 : (h < 53 ? 1 : 2);
        int wr = w < 49 ? 0 : (w < 53 ? 1 : 2);
        lab[t] = (dr * 3 + hr) * 3 + wr;
    }
    __syncthreads();

    // S = scale * (Q @ K^T) + bias + mask   (K=32 -> one WMMA per 16x16 tile)
    int rowOff = (lane >> 4) << 3, colL = lane & 15;
    for (int tile = wave; tile < 49; tile += 8) {
        int ti = tile / 7, tj = tile % 7;
        v8f acc = {};
        v16bf a = load_frag(Qs, HD, ti * 16, 0);
        v16bf b = load_frag(Ks, HD, tj * 16, 0);
        acc = wmma_bf16(a, b, acc);
        int col = tj * 16 + colL;
#pragma unroll
        for (int j = 0; j < 8; ++j) {
            int row = ti * 16 + rowOff + j;
            float m = (lab[row] != lab[col]) ? -100.0f : 0.0f;
            S[row * NPAD + col] = acc[j] * SCALE_Q
                                + biasPad[((size_t)hh * NPAD + row) * NPAD + col] + m;
        }
    }
    __syncthreads();

    // softmax: wave per row, 3 passes, no per-thread arrays
    for (int r = wave; r < NPAD; r += 8) {
        const float* srow = S + r * NPAD;
        float mx = -3.4e38f;
        for (int c = lane; c < NPAD; c += 32) mx = fmaxf(mx, srow[c]);
#pragma unroll
        for (int o = 16; o; o >>= 1) mx = fmaxf(mx, __shfl_xor(mx, o, 32));
        float sum = 0.f;
        for (int c = lane; c < NPAD; c += 32) sum += __expf(srow[c] - mx);
#pragma unroll
        for (int o = 16; o; o >>= 1) sum += __shfl_xor(sum, o, 32);
        float inv = 1.0f / sum;
        __bf16* prow = P + r * 128;
        for (int c = lane; c < 128; c += 32)
            prow[c] = (__bf16)(c < NPAD ? __expf(srow[c] - mx) * inv : 0.0f);
    }
    __syncthreads();

    // O = P @ V : 7(M) x 2(N) tiles, K-loop 128/32
    for (int tile = wave; tile < 14; tile += 8) {
        int mi = tile >> 1, ni = tile & 1;
        v8f acc = {};
#pragma unroll
        for (int kk = 0; kk < 4; ++kk) {
            v16bf a = load_frag(P, 128, mi * 16, kk * 32);
            v16bf b = load_frag(Vt, 128, ni * 16, kk * 32);
            acc = wmma_bf16(a, b, acc);
        }
#pragma unroll
        for (int j = 0; j < 8; ++j) {
            size_t row = (size_t)wb * NPAD + mi * 16 + rowOff + j;
            int col = hh * HD + ni * 16 + colL;
            attnOut[row * C + col] = (__bf16)acc[j];
        }
    }
}

// ---------------------------------------------------------------------------
// Host launch
// ---------------------------------------------------------------------------
extern "C" void kernel_launch(void* const* d_in, const int* in_sizes, int n_in,
                              void* d_out, int out_size, void* d_ws, size_t ws_size,
                              hipStream_t stream) {
    const float* x       = (const float*)d_in[0];
    const float* norm1_g = (const float*)d_in[1];
    const float* norm1_b = (const float*)d_in[2];
    const float* qkv_w   = (const float*)d_in[3];
    const float* qkv_b   = (const float*)d_in[4];
    const float* rpb     = (const float*)d_in[5];
    const float* proj_w  = (const float*)d_in[6];
    const float* proj_b  = (const float*)d_in[7];
    const float* norm2_g = (const float*)d_in[8];
    const float* norm2_b = (const float*)d_in[9];
    const float* fc1_w   = (const float*)d_in[10];
    const float* fc1_b   = (const float*)d_in[11];
    const float* fc2_w   = (const float*)d_in[12];
    const float* fc2_b   = (const float*)d_in[13];
    float* out = (float*)d_out;

    // workspace layout
    char* ws = (char*)d_ws;
    size_t off = 0;
    __bf16* wQkv  = (__bf16*)(ws + off); off += (size_t)QKVC * C * 2;
    __bf16* wProj = (__bf16*)(ws + off); off += (size_t)C * C * 2;
    __bf16* wFc1  = (__bf16*)(ws + off); off += (size_t)CH * C * 2;
    __bf16* wFc2  = (__bf16*)(ws + off); off += (size_t)C * CH * 2;
    float*  biasP = (float*)(ws + off);  off += (size_t)HEADS * NPAD * NPAD * 4;
    __bf16* bufA  = (__bf16*)(ws + off); off += (size_t)MROWS * C * 2;       // 88 MB
    __bf16* bufB  = (__bf16*)(ws + off); off += (size_t)TOKENS * CH * 2;     // 308 MB

    __bf16* Xw      = bufA;
    __bf16* QKVbuf  = bufB;
    __bf16* attnOut = bufA;   // Xw dead after QKV GEMM
    __bf16* yBuf    = bufA;   // attnOut dead after proj
    __bf16* h1Buf   = bufB;   // QKV dead after attention

    k_convert_bf16<<<1024, 256, 0, stream>>>(qkv_w,  wQkv,  QKVC * C);
    k_convert_bf16<<<1024, 256, 0, stream>>>(proj_w, wProj, C * C);
    k_convert_bf16<<<1024, 256, 0, stream>>>(fc1_w,  wFc1,  CH * C);
    k_convert_bf16<<<1024, 256, 0, stream>>>(fc2_w,  wFc2,  C * CH);

    k_bias_pad<<<(HEADS * NPAD * NPAD + 255) / 256, 256, 0, stream>>>(rpb, biasP);

    k_ln1_partition<<<MROWS / 8, 256, 0, stream>>>(x, norm1_g, norm1_b, Xw);

    k_gemm_qkv<<<(MROWS / BM) * (QKVC / BN), 256, 0, stream>>>(Xw, wQkv, qkv_b, QKVbuf,
                                                               MROWS, QKVC, C);

    size_t attnLds = (size_t)NPAD * HD * 2 * 2
                   + (size_t)HD * 128 * 2
                   + (size_t)NPAD * NPAD * 4
                   + (size_t)NPAD * 128 * 2
                   + NPAD * 4;
    k_attention<<<NWIN * HEADS, 256, attnLds, stream>>>(QKVbuf, biasP, attnOut);

    k_gemm_proj<<<(MROWS / BM) * (C / BN), 256, 0, stream>>>(attnOut, wProj, proj_b,
                                                             x, out, MROWS, C, C);

    k_ln2<<<TOKENS / 8, 256, 0, stream>>>(out, norm2_g, norm2_b, yBuf);

    k_gemm_fc1<<<(TOKENS / BM) * (CH / BN), 256, 0, stream>>>(yBuf, wFc1, fc1_b, h1Buf,
                                                              TOKENS, CH, C);

    k_gemm_fc2<<<(TOKENS / BM) * (C / BN), 256, 0, stream>>>(h1Buf, wFc2, fc2_b, out,
                                                             TOKENS, C, CH);
}
